// LSTMModel_83090437309104
// MI455X (gfx1250) — matrix operand order repaired and checked by reading, not compiled, not run on hardware
//
#include <hip/hip_runtime.h>
#include <cstdint>

typedef unsigned short u16;
typedef unsigned int   u32;
typedef __attribute__((ext_vector_type(16))) __bf16 v16bf;
typedef __attribute__((ext_vector_type(8)))  float  v8f;
typedef int v4i_vs __attribute__((vector_size(16)));

// ---------------- constants (match reference) ----------------
#define BB 64
#define TT 512
#define VV 50000
#define DD 128
#define MM 512
#define GG 2048          // 4*M
#define VPAD 50048       // 64-aligned padded vocab
#define LSTM_BLOCKS 32   // persistent-kernel grid; must co-reside

// ---------------- CDNA5 async global->LDS staging ----------------
#if __has_builtin(__builtin_amdgcn_global_load_async_to_lds_b128)
#define HAVE_ASYNC_LDS 1
#endif

__device__ __forceinline__ void async_copy_b128(void* lds, const void* gptr) {
#ifdef HAVE_ASYNC_LDS
  __builtin_amdgcn_global_load_async_to_lds_b128(
      (__attribute__((address_space(1))) v4i_vs*)(void*)gptr,
      (__attribute__((address_space(3))) v4i_vs*)lds,
      0, 0);
#else
  *(uint4*)lds = *(const uint4*)gptr;
#endif
}
__device__ __forceinline__ void async_wait() {
#ifdef HAVE_ASYNC_LDS
#if __has_builtin(__builtin_amdgcn_s_wait_asynccnt)
  __builtin_amdgcn_s_wait_asynccnt(0);
#else
  asm volatile("s_wait_asynccnt 0x0" ::: "memory");
#endif
#endif
}

// ---------------- small helpers ----------------
__device__ __forceinline__ u16 f2bf(float x) {
  u32 u = __builtin_bit_cast(u32, x);
  u32 r = (u + 0x7FFFu + ((u >> 16) & 1u)) >> 16;  // RNE
  return (u16)r;
}
__device__ __forceinline__ float bf2f(u16 h) {
  return __builtin_bit_cast(float, (u32)h << 16);
}
__device__ __forceinline__ float sigmoidf(float x) {
  return 1.0f / (1.0f + __expf(-x));
}

union BF16Frag { v16bf v; uint4 q[2]; };

__device__ __forceinline__ v8f wmma_bf16(v16bf a, v16bf b, v8f c) {
  // v_wmma_f32_16x16x32_bf16: (neg_a, A, neg_b, B, c_mod, C, reuse_a, reuse_b)
  return __builtin_amdgcn_wmma_f32_16x16x32_bf16(false, a, false, b, (short)0, c,
                                                 false, false);
}

// A (16x32 bf16, M x K) fragment per ISA layout:
// lanes 0-15: m=lane, halves 0-7 -> k=0..7, halves 8-15 -> k=16..23
// lanes 16-31: m=lane-16, halves -> k=8..15 and k=24..31
__device__ __forceinline__ v16bf load_a_frag(const u16* base, int rowStride,
                                             int mt, int kk, int lane) {
  int m  = mt * 16 + (lane & 15);
  int kb = kk * 32 + ((lane & 16) ? 8 : 0);
  const u16* r = base + m * rowStride + kb;
  BF16Frag f;
  f.q[0] = *(const uint4*)(r);        // k = kb .. kb+7
  f.q[1] = *(const uint4*)(r + 16);   // k = kb+16 .. kb+23
  return f.v;
}

// B (32x16 bf16, K x N) fragment, weights stored transposed: Wt[n][K]
// lane n = lane&15; halves 0-7 -> k=kb..kb+7, 8-15 -> k=kb+16..kb+23 (lanes>=16 take kb+8)
__device__ __forceinline__ v16bf load_b_frag(const u16* WtCol0, int K,
                                             int kk, int lane) {
  int n  = lane & 15;
  int kb = kk * 32 + ((lane & 16) ? 8 : 0);
  const u16* r = WtCol0 + (size_t)n * K + kb;
  BF16Frag f;
  f.q[0] = *(const uint4*)(r);
  f.q[1] = *(const uint4*)(r + 16);
  return f.v;
}

// ---------------- grid-wide barrier (sense reversal) ----------------
__device__ __forceinline__ void grid_barrier(u32* counter, u32* gen, u32 nblocks) {
  __syncthreads();
  if (threadIdx.x == 0) {
    __threadfence();
    u32 g = __hip_atomic_load(gen, __ATOMIC_RELAXED, __HIP_MEMORY_SCOPE_AGENT);
    u32 a = __hip_atomic_fetch_add(counter, 1u, __ATOMIC_ACQ_REL, __HIP_MEMORY_SCOPE_AGENT);
    if (a == nblocks - 1u) {
      __hip_atomic_store(counter, 0u, __ATOMIC_RELAXED, __HIP_MEMORY_SCOPE_AGENT);
      __hip_atomic_store(gen, g + 1u, __ATOMIC_RELEASE, __HIP_MEMORY_SCOPE_AGENT);
    } else {
      while (__hip_atomic_load(gen, __ATOMIC_ACQUIRE, __HIP_MEMORY_SCOPE_AGENT) == g) {
        __builtin_amdgcn_s_sleep(8);
      }
    }
    __threadfence();
  }
  __syncthreads();
}

// ---------------- kernels ----------------

// embedding gather + fp32->bf16
__global__ void embed_kernel(const int* __restrict__ x, const float* __restrict__ emb,
                             u16* __restrict__ e, int n) {
  int idx = blockIdx.x * 256 + threadIdx.x;
  if (idx >= n) return;
  int r = idx >> 7, d = idx & 127;
  e[idx] = f2bf(emb[(size_t)x[r] * DD + d]);
}

// W[k][N] fp32  ->  Wt[n][K] bf16  (zero pad n >= N)
__global__ void tconv_kernel(const float* __restrict__ W, u16* __restrict__ Wt,
                             int K, int N, long total) {
  long idx = (long)blockIdx.x * 256 + threadIdx.x;
  if (idx >= total) return;
  int n = (int)(idx / K);
  int k = (int)(idx % K);
  Wt[idx] = (n < N) ? f2bf(W[(size_t)k * N + n]) : (u16)0;
}

// out[R x 2048] = A_bf16[R x K] @ Wt_bf16 + bias ; block tile 64x64
__global__ __launch_bounds__(256) void gemm_rc_kernel(
    const u16* __restrict__ A, const u16* __restrict__ Wt,
    const float* __restrict__ bias, float* __restrict__ out, int K) {
  __shared__ __align__(16) u16 As[64 * 512];
  int r0 = blockIdx.x * 64;
  int c0 = blockIdx.y * 64;
  // stage 64 consecutive rows of A (contiguous in global) via async DMA to LDS
  {
    const uint4* Ag = (const uint4*)(A + (size_t)r0 * K);
    uint4* Asq = (uint4*)As;
    int tot = 64 * K / 8;
    for (int i = threadIdx.x; i < tot; i += 256) async_copy_b128(Asq + i, Ag + i);
    async_wait();
  }
  __syncthreads();
  int w = threadIdx.x >> 5, lane = threadIdx.x & 31;
  int mt = w & 3, ng = w >> 2;                 // 2 col-tiles per wave
  int col0 = c0 + ng * 32, col1 = col0 + 16;
  v8f acc0 = {}, acc1 = {};
  for (int kk = 0; kk < K / 32; ++kk) {
    v16bf a  = load_a_frag(As, K, mt, kk, lane);
    v16bf b0 = load_b_frag(Wt + (size_t)col0 * K, K, kk, lane);
    v16bf b1 = load_b_frag(Wt + (size_t)col1 * K, K, kk, lane);
    acc0 = wmma_bf16(a, b0, acc0);
    acc1 = wmma_bf16(a, b1, acc1);
  }
  int n = lane & 15, mo = (lane & 16) ? 8 : 0;
  float bv0 = bias[col0 + n], bv1 = bias[col1 + n];
  for (int r = 0; r < 8; ++r) {
    int m = mt * 16 + r + mo;
    out[(size_t)(r0 + m) * GG + col0 + n] = acc0[r] + bv0;
    out[(size_t)(r0 + m) * GG + col1 + n] = acc1[r] + bv1;
  }
}

// Persistent LSTM scan. 32 blocks x 256 threads. Block b owns hidden units
// [b*16, b*16+16). U columns (i,f,g,o for those units) live in LDS all T steps.
__global__ __launch_bounds__(256) void lstm_scan_kernel(
    const float* __restrict__ xw,   // [B*T][2048], row = m*T + t
    const u16*   __restrict__ Ut,   // [2048][512] bf16 transposed
    u16* __restrict__ h0buf, u16* __restrict__ h1buf,  // [64][512] double buffer
    u16* __restrict__ hseq,         // [B*T][512] bf16 output sequence
    u32* __restrict__ bar, int Tlen) {
  __shared__ __align__(16) u16  Us[64 * 512];   // 64 gate-cols x K=512
  __shared__ __align__(16) u16  hs[64 * 512];   // h_{t-1}: 64 batch x 512
  __shared__ float zbuf[64 * 64];               // z for 4 gates x 16 units
  __shared__ float cst[64 * 16];                // cell state
  const int j0 = blockIdx.x * 16;

  // load U slab: col-local cl = gate*16 + jj  ->  global col gate*512 + j0 + jj
  for (int i = threadIdx.x; i < 64 * 64; i += 256) {
    int cl = i >> 6, c4 = i & 63;
    int gcol = (cl >> 4) * MM + j0 + (cl & 15);
    async_copy_b128(((uint4*)Us) + cl * 64 + c4,
                    ((const uint4*)(Ut + (size_t)gcol * MM)) + c4);
  }
  for (int i = threadIdx.x; i < 64 * 16; i += 256) cst[i] = 0.0f;
  async_wait();
  __syncthreads();

  const int w = threadIdx.x >> 5, lane = threadIdx.x & 31;
  const int mt = w >> 1, gp = (w & 1) * 2;       // gates gp, gp+1
  const int n = lane & 15, mo = (lane & 16) ? 8 : 0;

  for (int t = 0; t < Tlen; ++t) {
    const u16* hrd = (t & 1) ? h1buf : h0buf;
    u16*       hwr = (t & 1) ? h0buf : h1buf;
    // stage h_{t-1} via async DMA while we fetch the xw accumulator init
    for (int i = threadIdx.x; i < 4096; i += 256)
      async_copy_b128(((uint4*)hs) + i, ((const uint4*)hrd) + i);

    // C init = precomputed input projection xw[:, t, gate-cols]
    v8f acc0, acc1;
    for (int r = 0; r < 8; ++r) {
      int m = mt * 16 + r + mo;
      size_t row = (size_t)m * Tlen + t;
      acc0[r] = xw[row * GG + (gp * MM + j0 + n)];
      acc1[r] = xw[row * GG + ((gp + 1) * MM + j0 + n)];
      if (t + 1 < Tlen)  // warm L2 for next step's xw row
        __builtin_prefetch(&xw[((size_t)m * Tlen + t + 1) * GG + gp * MM + j0], 0, 1);
    }
    async_wait();
    __syncthreads();

    // z += h_{t-1} @ U   (K = 512 -> 16 wmma per tile)
    for (int kk = 0; kk < 16; ++kk) {
      v16bf a  = load_a_frag(hs, MM, mt, kk, lane);
      v16bf b0 = load_b_frag(Us + (size_t)(gp * 16) * MM, MM, kk, lane);
      v16bf b1 = load_b_frag(Us + (size_t)(gp * 16 + 16) * MM, MM, kk, lane);
      acc0 = wmma_bf16(a, b0, acc0);
      acc1 = wmma_bf16(a, b1, acc1);
    }
    for (int r = 0; r < 8; ++r) {
      int m = mt * 16 + r + mo;
      zbuf[m * 64 + gp * 16 + n]       = acc0[r];
      zbuf[m * 64 + (gp + 1) * 16 + n] = acc1[r];
    }
    __syncthreads();

    // fused gates: i,f,o = sigmoid ; g = relu ; c = f*c + i*g ; h = o*relu(c)
    for (int q = threadIdx.x; q < 1024; q += 256) {
      int m = q >> 4, jj = q & 15;
      float zi = zbuf[m * 64 + jj];
      float zf = zbuf[m * 64 + 16 + jj];
      float zg = zbuf[m * 64 + 32 + jj];
      float zo = zbuf[m * 64 + 48 + jj];
      float ig = sigmoidf(zi), fg = sigmoidf(zf), og = sigmoidf(zo);
      float gg = fmaxf(zg, 0.0f);
      float c  = fg * cst[m * 16 + jj] + ig * gg;
      cst[m * 16 + jj] = c;
      u16 hb = f2bf(og * fmaxf(c, 0.0f));
      hwr[m * MM + j0 + jj] = hb;
      hseq[((size_t)m * Tlen + t) * MM + j0 + jj] = hb;
    }
    __syncthreads();
    grid_barrier(bar, bar + 1, gridDim.x);
  }
}

// global max pool over T
__global__ void maxpool_kernel(const u16* __restrict__ h2, u16* __restrict__ pool) {
  int idx = blockIdx.x * 256 + threadIdx.x;
  if (idx >= BB * MM) return;
  int m = idx >> 9, j = idx & 511;
  const u16* base = h2 + (size_t)m * TT * MM + j;
  float mx = -3.402823466e38f;
  for (int t = 0; t < TT; ++t) mx = fmaxf(mx, bf2f(base[(size_t)t * MM]));
  pool[idx] = f2bf(mx);
}

// out[64 x N] = act(A_bf16[64x512] @ Wt + bias); writes f32 (+ optional bf16)
__global__ __launch_bounds__(256) void gemm64_kernel(
    const u16* __restrict__ A, const u16* __restrict__ Wt,
    const float* __restrict__ bias, float* __restrict__ outF,
    u16* __restrict__ outB, int Nreal, int NStride, int relu) {
  __shared__ __align__(16) u16 As[64 * 512];
  for (int i = threadIdx.x; i < 4096; i += 256)
    async_copy_b128(((uint4*)As) + i, ((const uint4*)A) + i);
  async_wait();
  __syncthreads();
  int c0 = blockIdx.x * 64;
  int w = threadIdx.x >> 5, lane = threadIdx.x & 31;
  int mt = w & 3, ng = w >> 2;
  int col0 = c0 + ng * 32, col1 = col0 + 16;
  v8f acc0 = {}, acc1 = {};
  for (int kk = 0; kk < 16; ++kk) {
    v16bf a  = load_a_frag(As, MM, mt, kk, lane);
    v16bf b0 = load_b_frag(Wt + (size_t)col0 * MM, MM, kk, lane);
    v16bf b1 = load_b_frag(Wt + (size_t)col1 * MM, MM, kk, lane);
    acc0 = wmma_bf16(a, b0, acc0);
    acc1 = wmma_bf16(a, b1, acc1);
  }
  int n = lane & 15, mo = (lane & 16) ? 8 : 0;
  float bv0 = (col0 + n) < Nreal ? bias[col0 + n] : 0.0f;
  float bv1 = (col1 + n) < Nreal ? bias[col1 + n] : 0.0f;
  for (int r = 0; r < 8; ++r) {
    int m = mt * 16 + r + mo;
    float v0 = acc0[r] + bv0, v1 = acc1[r] + bv1;
    if (relu) { v0 = fmaxf(v0, 0.0f); v1 = fmaxf(v1, 0.0f); }
    outF[(size_t)m * NStride + col0 + n] = v0;
    outF[(size_t)m * NStride + col1 + n] = v1;
    if (outB) {
      outB[(size_t)m * NStride + col0 + n] = f2bf(v0);
      outB[(size_t)m * NStride + col1 + n] = f2bf(v1);
    }
  }
}

// one block per batch row
__global__ __launch_bounds__(256) void softmax_kernel(
    const float* __restrict__ logits, float* __restrict__ out, int V, int stride) {
  __shared__ float red[256];
  const float* row = logits + (size_t)blockIdx.x * stride;
  float mx = -3.402823466e38f;
  for (int j = threadIdx.x; j < V; j += 256) mx = fmaxf(mx, row[j]);
  red[threadIdx.x] = mx; __syncthreads();
  for (int s = 128; s > 0; s >>= 1) {
    if (threadIdx.x < s) red[threadIdx.x] = fmaxf(red[threadIdx.x], red[threadIdx.x + s]);
    __syncthreads();
  }
  mx = red[0]; __syncthreads();
  float sum = 0.0f;
  for (int j = threadIdx.x; j < V; j += 256) sum += __expf(row[j] - mx);
  red[threadIdx.x] = sum; __syncthreads();
  for (int s = 128; s > 0; s >>= 1) {
    if (threadIdx.x < s) red[threadIdx.x] += red[threadIdx.x + s];
    __syncthreads();
  }
  float inv = 1.0f / red[0];
  float* orow = out + (size_t)blockIdx.x * V;
  for (int j = threadIdx.x; j < V; j += 256) orow[j] = __expf(row[j] - mx) * inv;
}

// ---------------- launch ----------------
extern "C" void kernel_launch(void* const* d_in, const int* in_sizes, int n_in,
                              void* d_out, int out_size, void* d_ws, size_t ws_size,
                              hipStream_t stream) {
  const int*   x   = (const int*)d_in[0];
  const float* emb = (const float*)d_in[1];
  const float* W1  = (const float*)d_in[2];
  const float* U1  = (const float*)d_in[3];
  const float* b1  = (const float*)d_in[4];
  const float* W2  = (const float*)d_in[5];
  const float* U2  = (const float*)d_in[6];
  const float* b2  = (const float*)d_in[7];
  const float* Wd  = (const float*)d_in[8];
  const float* bd  = (const float*)d_in[9];
  const float* Wo  = (const float*)d_in[10];
  const float* bo  = (const float*)d_in[11];

  char* ws = (char*)d_ws;
  size_t off = 0;
  auto alloc = [&](size_t bytes) { size_t r = off; off = (off + bytes + 255) & ~(size_t)255; return r; };

  size_t o_bar  = alloc(256);                                  // barrier counter+gen
  size_t o_hst  = alloc((size_t)2 * BB * MM * 2);              // h double buffer
  size_t o_e    = alloc((size_t)BB * TT * DD * 2);             // embedded bf16
  size_t o_xw   = alloc((size_t)BB * TT * GG * 4);             // xw f32 (reused L1/L2)
  size_t o_h1   = alloc((size_t)BB * TT * MM * 2);
  size_t o_h2   = alloc((size_t)BB * TT * MM * 2);
  size_t o_W1t  = alloc((size_t)GG * DD * 2);
  size_t o_U1t  = alloc((size_t)GG * MM * 2);
  size_t o_W2t  = alloc((size_t)GG * MM * 2);
  size_t o_U2t  = alloc((size_t)GG * MM * 2);
  size_t o_Wdt  = alloc((size_t)MM * MM * 2);
  size_t o_Wot  = alloc((size_t)VPAD * MM * 2);
  size_t o_pool = alloc((size_t)BB * MM * 2);
  size_t o_dF   = alloc((size_t)BB * MM * 4);
  size_t o_dB   = alloc((size_t)BB * MM * 2);
  size_t o_log  = alloc((size_t)BB * VPAD * 4);

  u32*   bar   = (u32*)(ws + o_bar);
  u16*   hA    = (u16*)(ws + o_hst);
  u16*   hB    = hA + (size_t)BB * MM;
  u16*   e_bf  = (u16*)(ws + o_e);
  float* xw    = (float*)(ws + o_xw);
  u16*   h1    = (u16*)(ws + o_h1);
  u16*   h2    = (u16*)(ws + o_h2);
  u16*   W1t   = (u16*)(ws + o_W1t);
  u16*   U1t   = (u16*)(ws + o_U1t);
  u16*   W2t   = (u16*)(ws + o_W2t);
  u16*   U2t   = (u16*)(ws + o_U2t);
  u16*   Wdt   = (u16*)(ws + o_Wdt);
  u16*   Wot   = (u16*)(ws + o_Wot);
  u16*   pool  = (u16*)(ws + o_pool);
  float* dF    = (float*)(ws + o_dF);
  u16*   dB    = (u16*)(ws + o_dB);
  float* logit = (float*)(ws + o_log);

  (void)hipMemsetAsync(bar, 0, 256, stream);

  // weight transpose+convert (one-time per call; bandwidth-trivial)
  auto tc = [&](const float* W, u16* Wt, int K, int N, long total) {
    tconv_kernel<<<(int)((total + 255) / 256), 256, 0, stream>>>(W, Wt, K, N, total);
  };
  tc(W1, W1t, DD, GG, (long)GG * DD);
  tc(U1, U1t, MM, GG, (long)GG * MM);
  tc(W2, W2t, MM, GG, (long)GG * MM);
  tc(U2, U2t, MM, GG, (long)GG * MM);
  tc(Wd, Wdt, MM, MM, (long)MM * MM);
  tc(Wo, Wot, MM, VV, (long)VPAD * MM);

  embed_kernel<<<(BB * TT * DD + 255) / 256, 256, 0, stream>>>(x, emb, e_bf, BB * TT * DD);

  // layer 1
  gemm_rc_kernel<<<dim3(BB * TT / 64, GG / 64), 256, 0, stream>>>(e_bf, W1t, b1, xw, DD);
  (void)hipMemsetAsync(hA, 0, (size_t)2 * BB * MM * 2, stream);
  lstm_scan_kernel<<<LSTM_BLOCKS, 256, 0, stream>>>(xw, U1t, hA, hB, h1, bar, TT);

  // layer 2
  gemm_rc_kernel<<<dim3(BB * TT / 64, GG / 64), 256, 0, stream>>>(h1, W2t, b2, xw, MM);
  (void)hipMemsetAsync(hA, 0, (size_t)2 * BB * MM * 2, stream);
  lstm_scan_kernel<<<LSTM_BLOCKS, 256, 0, stream>>>(xw, U2t, hA, hB, h2, bar, TT);

  // pool -> dense(relu) -> logits -> softmax
  maxpool_kernel<<<(BB * MM + 255) / 256, 256, 0, stream>>>(h2, pool);
  gemm64_kernel<<<MM / 64, 256, 0, stream>>>(pool, Wdt, bd, dF, dB, MM, MM, 1);
  gemm64_kernel<<<VPAD / 64, 256, 0, stream>>>(dB, Wot, bo, logit, (u16*)nullptr, VV, VPAD, 0);
  softmax_kernel<<<BB, 256, 0, stream>>>(logit, (float*)d_out, VV, VPAD);
}
